// DualEncoder_62749472195477
// MI455X (gfx1250) — compile-verified, matrix operand order checked by reading, not run on hardware
//
#include <hip/hip_runtime.h>
#include <hip/hip_bf16.h>

// ---------------------------------------------------------------------------
// Dual transformer encoder for MI455X (gfx1250, wave32, WMMA + TDM).
// All matmuls run on v_wmma_f32_16x16x32_bf16 (f32 accumulate). Weights are
// pre-converted once per launch to bf16, pre-transposed to NxK so the Tensor
// Data Mover can DMA B-tiles into LDS in fragment layout with no transpose.
// ---------------------------------------------------------------------------

#define B_  8
#define T_  1024
#define MT  8192          // B*T rows
#define D_  512
#define H_  8
#define DK_ 64
#define L_  6
#define DFF_ 2048
#define IDIM_ 80
#define CHUNK_ 16
#define LEFT_ 4

typedef __attribute__((ext_vector_type(16))) __bf16 v16bf;
typedef __attribute__((ext_vector_type(8)))  float  v8f;
typedef __attribute__((ext_vector_type(4)))  unsigned uint32x4;
typedef __attribute__((ext_vector_type(8)))  int      int32x8;
typedef __attribute__((ext_vector_type(4)))  int      int32x4;

#define WMMA_BF16(a, b, c) \
  __builtin_amdgcn_wmma_f32_16x16x32_bf16(false, (a), false, (b), (short)0, (c), false, false)

__device__ __forceinline__ v8f zero8() {
  v8f z = {0.f, 0.f, 0.f, 0.f, 0.f, 0.f, 0.f, 0.f};
  return z;
}

// f32 -> bf16 round-to-nearest-even
__device__ __forceinline__ __bf16 f2bf(float f) {
  union { float f; unsigned u; } x{f};
  unsigned r = x.u + 0x7FFFu + ((x.u >> 16) & 1u);
  unsigned short hs = (unsigned short)(r >> 16);
  union { unsigned short s; __bf16 b; } y{hs};
  return y.b;
}

// pack two f32 into one dword of 2 bf16 (RNE)
__device__ __forceinline__ unsigned pack2bf(float a, float b) {
  union { float f; unsigned u; } xa{a}, xb{b};
  unsigned ra = xa.u + 0x7FFFu + ((xa.u >> 16) & 1u);
  unsigned rb = xb.u + 0x7FFFu + ((xb.u >> 16) & 1u);
  return (ra >> 16) | (rb & 0xFFFF0000u);
}

// Load one 16x32 bf16 WMMA operand fragment from LDS.
// ISA 7.12.2 16-bit layout: lane&15 selects row/col, lane>>4 the K-half.
__device__ __forceinline__ v16bf ldfrag(const __bf16* p, int stride) {
  const int lane = threadIdx.x & 31;
  const int idx  = lane & 15;
  const int koff = (lane >> 4) << 3;
  const __bf16* q = p + idx * stride + koff;
  union { v16bf v; uint4 u[2]; } r;
  r.u[0] = *(const uint4*)(q);
  r.u[1] = *(const uint4*)(q + 16);
  return r.v;
}

// ---------------------------------------------------------------------------
// C = epilogue(A[MxK] @ W + bias [+resid]); W passed as bf16, TRANSPOSED (NxK).
// Block: 256 threads (8 waves); tile 128x128x32; wave tile 32x64 (2x4 WMMA).
// B-tiles (128x32 bf16, contiguous-in-K rows) are fetched by the TDM straight
// into LDS; tensor_dim0=K zero-fills any K tail (e.g. embedding K=80).
// ---------------------------------------------------------------------------
__global__ __launch_bounds__(256) void gemm_wmma_kernel(
    const float* __restrict__ A, const __bf16* __restrict__ Wt,
    const float* __restrict__ bias, const float* __restrict__ resid,
    float* __restrict__ C, int M, int K, int N, int relu) {
  __shared__ __bf16 sA[128 * 32];   // row-major, stride 32
  __shared__ __bf16 sB[128 * 32];   // col n contiguous in k, stride 32

  const int tid  = threadIdx.x;
  const int lane = tid & 31;
  const int wave = tid >> 5;
  const int wm = wave & 3;
  const int wn = wave >> 2;
  const int m0 = blockIdx.y * 128;
  const int n0 = blockIdx.x * 128;

  v8f acc[2][4];
#pragma unroll
  for (int mi = 0; mi < 2; ++mi)
#pragma unroll
    for (int ni = 0; ni < 4; ++ni) acc[mi][ni] = zero8();

  // Invariant words of the TDM descriptor (D# group 1, ISA 8.4):
  // data_size=1 (2B), tensor_dim0=K, tensor_dim1=N, tile 32x128, stride0=K.
  const unsigned g1w1 = ((unsigned)K & 0xFFFFu) << 16;
  const unsigned g1w2 = ((unsigned)K >> 16) | (((unsigned)N & 0xFFFFu) << 16);
  const unsigned g1w3 = ((unsigned)N >> 16) | (32u << 16);
  const unsigned ldsb = (unsigned)(unsigned long long)(&sB[0]);

  for (int k0 = 0; k0 < K; k0 += 32) {
    __syncthreads();

    // ---- B tile via Tensor Data Mover (one wave issues; TENSORcnt) ----
    if (wave == 0) {
      unsigned long long ga =
          (unsigned long long)(const void*)(Wt + (size_t)n0 * K + k0);
      uint32x4 g0 = {1u,                                   // count=1
                     ldsb,                                 // lds_addr
                     (unsigned)ga,                         // global_addr lo
                     (((unsigned)(ga >> 32)) & 0x01FFFFFFu) | 0x80000000u}; // hi | type=2
      int32x8 g1 = {(int)0x00010000u, (int)g1w1, (int)g1w2, (int)g1w3,
                    (int)128u,        (int)(unsigned)K, 0, 0};
      int32x4 gz4 = {0, 0, 0, 0};
      int32x8 gz8 = {0, 0, 0, 0, 0, 0, 0, 0};
      __builtin_amdgcn_tensor_load_to_lds(g0, g1, gz4, gz4, gz8, 0);
      __builtin_amdgcn_s_wait_tensorcnt(0);
    }

    // ---- A tile 128x32: vectorized fast path when tile fully in-bounds ----
    if (k0 + 32 <= K) {
#pragma unroll
      for (int it = 0; it < 4; ++it) {
        int lin = it * 256 + tid;                 // float4 index, 0..1023
        int r = lin >> 3, c4 = lin & 7;
        const float4 va = *(const float4*)&A[(size_t)(m0 + r) * K + k0 + c4 * 4];
        uint2 p; p.x = pack2bf(va.x, va.y); p.y = pack2bf(va.z, va.w);
        *(uint2*)&sA[r * 32 + c4 * 4] = p;
      }
    } else {
#pragma unroll
      for (int i = 0; i < 16; ++i) {
        int lin = i * 256 + tid;
        int r = lin >> 5, c = lin & 31;
        float v = (k0 + c < K) ? A[(size_t)(m0 + r) * K + (k0 + c)] : 0.f;
        sA[r * 32 + c] = f2bf(v);
      }
    }
    if (k0 + 32 < K)
      __builtin_prefetch(&A[(size_t)(m0 + (tid >> 1)) * K + (k0 + 32)], 0, 1);
    __syncthreads();

    v16bf af[2], bfg[4];
#pragma unroll
    for (int mi = 0; mi < 2; ++mi)
      af[mi] = ldfrag(sA + (wm * 32 + mi * 16) * 32, 32);
#pragma unroll
    for (int ni = 0; ni < 4; ++ni)
      bfg[ni] = ldfrag(sB + (wn * 64 + ni * 16) * 32, 32);
#pragma unroll
    for (int mi = 0; mi < 2; ++mi)
#pragma unroll
      for (int ni = 0; ni < 4; ++ni)
        acc[mi][ni] = WMMA_BF16(af[mi], bfg[ni], acc[mi][ni]);
  }

  const int row0 = (lane >> 4) << 3;
  const int coll = lane & 15;
#pragma unroll
  for (int mi = 0; mi < 2; ++mi) {
#pragma unroll
    for (int r = 0; r < 8; ++r) {
      int grow = m0 + wm * 32 + mi * 16 + row0 + r;
#pragma unroll
      for (int ni = 0; ni < 4; ++ni) {
        int gcol = n0 + wn * 64 + ni * 16 + coll;
        float v = acc[mi][ni][r];
        if (bias) v += bias[gcol];
        if (relu) v = fmaxf(v, 0.f);
        if (resid) v += resid[(size_t)grow * N + gcol];
        C[(size_t)grow * N + gcol] = v;
      }
    }
  }
}

// ---------------------------------------------------------------------------
// Flash-style attention, one wave per (batch, head, 16-row query chunk).
// ---------------------------------------------------------------------------
__global__ __launch_bounds__(32) void attn_wmma_kernel(
    const float* __restrict__ Q, const float* __restrict__ K,
    const float* __restrict__ V, float* __restrict__ O, int mode) {
  __shared__ __bf16 sQ[16 * 64];
  __shared__ __bf16 sK[32 * 64];
  __shared__ __bf16 sVt[64 * 32];
  __shared__ float  sS[16 * 32];
  __shared__ __bf16 sP[16 * 32];
  __shared__ float  sAlpha[16];
  __shared__ float  sLinv[16];

  const int lane  = threadIdx.x;
  const int qt    = blockIdx.x;
  const int hh    = blockIdx.y;
  const int bb    = blockIdx.z;
  const int qbase = qt * 16;
  const float scale = 0.125f;   // 1/sqrt(DK)

  // stage Q (pre-scaled), float4 loads
#pragma unroll
  for (int i = 0; i < 8; ++i) {
    int lin = i * 32 + lane;                  // float4 index over 16x64
    int r = lin >> 4, d0 = (lin & 15) * 4;
    const float4 qv = *(const float4*)&Q[((size_t)(bb * T_ + qbase + r)) * D_ + hh * DK_ + d0];
    uint2 p; p.x = pack2bf(qv.x * scale, qv.y * scale);
    p.y = pack2bf(qv.z * scale, qv.w * scale);
    *(uint2*)&sQ[r * 64 + d0] = p;
  }
  __syncthreads();
  v16bf qf0 = ldfrag(sQ, 64);
  v16bf qf1 = ldfrag(sQ + 32, 64);

  v8f oacc[4];
#pragma unroll
  for (int j = 0; j < 4; ++j) oacc[j] = zero8();
  float mrow = -3.0e38f, lrow = 0.0f;

  int kt0 = 0, kt1 = T_ / 32;
  if (mode) {
    int c0 = qt - LEFT_; if (c0 < 0) c0 = 0;
    kt0 = (c0 * CHUNK_) >> 5;
    kt1 = ((qt * CHUNK_ + CHUNK_ - 1) >> 5) + 1;
  }

  for (int kt = kt0; kt < kt1; ++kt) {
    __syncthreads();
    // stage 32 keys x 64 dims of K (packed) and V (transposed)
#pragma unroll
    for (int i = 0; i < 16; ++i) {
      int lin = i * 32 + lane;                // float4 index over 32x64
      int kk = lin >> 4, d0 = (lin & 15) * 4;
      size_t g = ((size_t)(bb * T_ + kt * 32 + kk)) * D_ + hh * DK_ + d0;
      const float4 kv = *(const float4*)&K[g];
      const float4 vv = *(const float4*)&V[g];
      uint2 p; p.x = pack2bf(kv.x, kv.y); p.y = pack2bf(kv.z, kv.w);
      *(uint2*)&sK[kk * 64 + d0] = p;
      sVt[(d0 + 0) * 32 + kk] = f2bf(vv.x);
      sVt[(d0 + 1) * 32 + kk] = f2bf(vv.y);
      sVt[(d0 + 2) * 32 + kk] = f2bf(vv.z);
      sVt[(d0 + 3) * 32 + kk] = f2bf(vv.w);
    }
    __syncthreads();

    v8f s0 = zero8(), s1 = zero8();
#pragma unroll
    for (int s = 0; s < 2; ++s) {
      v16bf a  = (s == 0) ? qf0 : qf1;
      v16bf b0 = ldfrag(sK + 0 * 64 + s * 32, 64);
      v16bf b1 = ldfrag(sK + 16 * 64 + s * 32, 64);
      s0 = WMMA_BF16(a, b0, s0);
      s1 = WMMA_BF16(a, b1, s1);
    }

    {
      int r0 = (lane >> 4) << 3;
      int c  = lane & 15;
#pragma unroll
      for (int r = 0; r < 8; ++r) {
        sS[(r0 + r) * 32 + c]      = s0[r];
        sS[(r0 + r) * 32 + c + 16] = s1[r];
      }
    }
    __syncthreads();

    if (lane < 16) {
      float sv[32];
      float tmax = -3.0e38f;
#pragma unroll
      for (int c = 0; c < 32; ++c) {
        float val = sS[lane * 32 + c];
        int kc = (kt * 32 + c) >> 4;
        bool ok = (mode == 0) || (kc >= qt - LEFT_ && kc <= qt);
        val = ok ? val : -3.0e38f;
        sv[c] = val;
        tmax = fmaxf(tmax, val);
      }
      float nm    = fmaxf(mrow, tmax);
      float alpha = __expf(mrow - nm);
      float psum  = 0.f;
#pragma unroll
      for (int c = 0; c < 32; ++c) {
        float p = __expf(sv[c] - nm);
        psum += p;
        sP[lane * 32 + c] = f2bf(p);
      }
      lrow = lrow * alpha + psum;
      mrow = nm;
      sAlpha[lane] = alpha;
    }
    __syncthreads();

    {
      int r0 = (lane >> 4) << 3;
#pragma unroll
      for (int r = 0; r < 8; ++r) {
        float a = sAlpha[r0 + r];
#pragma unroll
        for (int j = 0; j < 4; ++j) oacc[j][r] *= a;
      }
    }
    v16bf pf = ldfrag(sP, 32);
#pragma unroll
    for (int j = 0; j < 4; ++j) {
      v16bf vf = ldfrag(sVt + (j * 16) * 32, 32);
      oacc[j] = WMMA_BF16(pf, vf, oacc[j]);
    }
  }

  if (lane < 16) sLinv[lane] = (lrow > 0.f) ? (1.0f / lrow) : 0.f;
  __syncthreads();

  {
    int r0 = (lane >> 4) << 3;
    int c  = lane & 15;
#pragma unroll
    for (int r = 0; r < 8; ++r) {
      float inv = sLinv[r0 + r];
      size_t base = ((size_t)(bb * T_ + qbase + r0 + r)) * D_ + hh * DK_;
#pragma unroll
      for (int j = 0; j < 4; ++j)
        O[base + j * 16 + c] = oacc[j][r] * inv;
    }
  }
}

// ---------------------------------------------------------------------------
// LayerNorm over D=512: one wave per row (8 rows / 256-thread block).
// ---------------------------------------------------------------------------
__global__ __launch_bounds__(256) void ln_kernel(
    const float* __restrict__ x, const float* __restrict__ g,
    const float* __restrict__ b, float* __restrict__ y, int relu) {
  const int lane = threadIdx.x & 31;
  const int wave = threadIdx.x >> 5;
  const size_t row = (size_t)blockIdx.x * 8 + wave;
  const float* xr = x + row * D_;
  float vals[16];
  float s = 0.f;
#pragma unroll
  for (int j = 0; j < 16; ++j) { vals[j] = xr[lane + j * 32]; s += vals[j]; }
#pragma unroll
  for (int o = 16; o > 0; o >>= 1) s += __shfl_xor(s, o, 32);
  float mean = s * (1.0f / 512.0f);
  float vs = 0.f;
#pragma unroll
  for (int j = 0; j < 16; ++j) { float d = vals[j] - mean; vs += d * d; }
#pragma unroll
  for (int o = 16; o > 0; o >>= 1) vs += __shfl_xor(vs, o, 32);
  float inv = rsqrtf(vs * (1.0f / 512.0f) + 1e-12f);
  float* yr = y + row * D_;
#pragma unroll
  for (int j = 0; j < 16; ++j) {
    int c = lane + j * 32;
    float v = (vals[j] - mean) * inv * g[c] + b[c];
    if (relu) v = fmaxf(v, 0.f);
    yr[c] = v;
  }
}

// x = x*sqrt(D) + sinusoidal positional encoding
__global__ void posenc_kernel(float* __restrict__ x) {
  size_t i = (size_t)blockIdx.x * blockDim.x + threadIdx.x;
  int d = (int)(i & (D_ - 1));
  int t = (int)((i >> 9) & (T_ - 1));
  float div = __expf((float)(d & ~1) * (-9.210340371976184f / 512.0f));
  float ang = (float)t * div;
  float pe = (d & 1) ? __cosf(ang) : __sinf(ang);
  x[i] = x[i] * 22.627416997969522f + pe;
}

// out[n*K + k] = bf16(in[k*N + n])  (one-time weight transpose+convert)
__global__ void wconv_kernel(const float* __restrict__ in, __bf16* __restrict__ out,
                             int K, int N) {
  size_t i = (size_t)blockIdx.x * blockDim.x + threadIdx.x;
  if (i >= (size_t)K * N) return;
  int n = (int)(i / K);
  int k = (int)(i - (size_t)n * K);
  out[i] = f2bf(in[(size_t)k * N + n]);
}

__global__ void masks_out_kernel(const unsigned char* __restrict__ m,
                                 float* __restrict__ o, int n) {
  int i = blockIdx.x * blockDim.x + threadIdx.x;
  if (i < n) o[i] = m[i] ? 1.0f : 0.0f;
}

// ---------------------------------------------------------------------------
// Host orchestration
// ---------------------------------------------------------------------------
static void launch_gemm(const float* A, const __bf16* Wt, const float* bias,
                        const float* resid, float* C, int M, int K, int N,
                        int relu, hipStream_t s) {
  dim3 grid(N / 128, M / 128);
  gemm_wmma_kernel<<<grid, 256, 0, s>>>(A, Wt, bias, resid, C, M, K, N, relu);
}

static void launch_wconv(const float* in, __bf16* out, int K, int N, hipStream_t s) {
  size_t tot = (size_t)K * N;
  wconv_kernel<<<(unsigned)((tot + 255) / 256), 256, 0, s>>>(in, out, K, N);
}

extern "C" void kernel_launch(void* const* d_in, const int* in_sizes, int n_in,
                              void* d_out, int out_size, void* d_ws, size_t ws_size,
                              hipStream_t stream) {
  const float* xs      = (const float*)d_in[0];
  const unsigned char* masks = (const unsigned char*)d_in[1];
  const float* emb_w   = (const float*)d_in[2];
  const float* emb_b   = (const float*)d_in[3];
  const float* emb_g   = (const float*)d_in[4];
  const float* emb_be  = (const float*)d_in[5];
  const float* wq      = (const float*)d_in[6];
  const float* bq      = (const float*)d_in[7];
  const float* wk      = (const float*)d_in[8];
  const float* bk      = (const float*)d_in[9];
  const float* wv      = (const float*)d_in[10];
  const float* bv      = (const float*)d_in[11];
  const float* wo      = (const float*)d_in[12];
  const float* bo      = (const float*)d_in[13];
  const float* ln1_g   = (const float*)d_in[14];
  const float* ln1_b   = (const float*)d_in[15];
  const float* ln2_g   = (const float*)d_in[16];
  const float* ln2_b   = (const float*)d_in[17];
  const float* ff1_w   = (const float*)d_in[18];
  const float* ff1_b   = (const float*)d_in[19];
  const float* ff2_w   = (const float*)d_in[20];
  const float* ff2_b   = (const float*)d_in[21];
  const float* fin_g   = (const float*)d_in[22];
  const float* fin_b   = (const float*)d_in[23];

  const size_t NX = (size_t)MT * D_;
  float* ws  = (float*)d_ws;
  float* x0  = ws;
  float* x   = x0 + NX;
  float* t1  = x  + NX;
  float* h   = t1 + NX;
  float* q   = h  + NX;
  float* kbf = q  + NX;
  float* vbf = kbf + NX;
  float* ao  = vbf + NX;
  float* mid = ao + NX;                                    // MT x DFF
  __bf16* wtb = (__bf16*)(mid + (size_t)MT * DFF_);        // converted weights

  // bf16 transposed weight layout in ws
  const size_t SQ = (size_t)D_ * D_;        // 262144
  const size_t SF = (size_t)D_ * DFF_;      // 1048576
  const size_t PER_L = 4 * SQ + 2 * SF;
  __bf16* embT = wtb;                                      // 512 x 80
  __bf16* lw   = wtb + (size_t)D_ * IDIM_;

  // One-time per launch: convert+transpose all weights to bf16 NxK
  launch_wconv(emb_w, embT, IDIM_, D_, stream);
  for (int l = 0; l < L_; ++l) {
    __bf16* base = lw + (size_t)l * PER_L;
    launch_wconv(wq + l * SQ, base + 0 * SQ, D_, D_, stream);
    launch_wconv(wk + l * SQ, base + 1 * SQ, D_, D_, stream);
    launch_wconv(wv + l * SQ, base + 2 * SQ, D_, D_, stream);
    launch_wconv(wo + l * SQ, base + 3 * SQ, D_, D_, stream);
    launch_wconv(ff1_w + l * SF, base + 4 * SQ, D_, DFF_, stream);
    launch_wconv(ff2_w + l * SF, base + 4 * SQ + SF, DFF_, D_, stream);
  }

  float* outp = (float*)d_out;

  // Embedding: xs @ emb_w + emb_b -> LN(+ReLU) -> posenc
  launch_gemm(xs, embT, emb_b, nullptr, t1, MT, IDIM_, D_, 0, stream);
  ln_kernel<<<MT / 8, 256, 0, stream>>>(t1, emb_g, emb_be, x0, 1);
  posenc_kernel<<<(int)(NX / 256), 256, 0, stream>>>(x0);

  for (int mode = 0; mode < 2; ++mode) {
    (void)hipMemcpyAsync(x, x0, NX * sizeof(float), hipMemcpyDeviceToDevice, stream);
    for (int l = 0; l < L_; ++l) {
      __bf16* base = lw + (size_t)l * PER_L;
      const size_t bo_off = (size_t)l * D_;
      // x = x + O( attn( LN1(x) ) )
      ln_kernel<<<MT / 8, 256, 0, stream>>>(x, ln1_g + bo_off, ln1_b + bo_off, h, 0);
      launch_gemm(h, base + 0 * SQ, bq + bo_off, nullptr, q,   MT, D_, D_, 0, stream);
      launch_gemm(h, base + 1 * SQ, bk + bo_off, nullptr, kbf, MT, D_, D_, 0, stream);
      launch_gemm(h, base + 2 * SQ, bv + bo_off, nullptr, vbf, MT, D_, D_, 0, stream);
      attn_wmma_kernel<<<dim3(T_ / 16, H_, B_), 32, 0, stream>>>(q, kbf, vbf, ao, mode);
      launch_gemm(ao, base + 3 * SQ, bo + bo_off, x, t1, MT, D_, D_, 0, stream);
      // x = t1 + FFN( LN2(t1) )
      ln_kernel<<<MT / 8, 256, 0, stream>>>(t1, ln2_g + bo_off, ln2_b + bo_off, h, 0);
      launch_gemm(h, base + 4 * SQ, ff1_b + (size_t)l * DFF_, nullptr, mid,
                  MT, D_, DFF_, 1, stream);
      launch_gemm(mid, base + 4 * SQ + SF, ff2_b + bo_off, t1, x,
                  MT, DFF_, D_, 0, stream);
    }
    ln_kernel<<<MT / 8, 256, 0, stream>>>(x, fin_g, fin_b, outp + (size_t)mode * NX, 0);
  }
  masks_out_kernel<<<(B_ * T_ + 255) / 256, 256, 0, stream>>>(masks, outp + 2 * NX, B_ * T_);
}
